// GumbelVectorQuantizer_55336358642373
// MI455X (gfx1250) — compile-verified, matrix operand order checked by reading, not run on hardware
//
#include <hip/hip_runtime.h>
#include <hip/hip_bf16.h>

// ---------------------------------------------------------------------------
// Gumbel Vector Quantizer forward for MI455X (gfx1250, wave32, WMMA).
//
// logits = x @ W + b        : 65536x512 @ 512x640, f16 WMMA w/ f32 accum
// k      = argmax_g(logits) : wave32 shfl reductions
// out    = codebook[g, k]   : L2-resident gather
// perplexities             : global 640-entry accumulators + tiny finalize
//
// HBM floor: read x (128MiB) + write out (64MiB) ~= 201MB -> ~8.6us @23.3TB/s.
// M_TILE=32 + 2-way B-fragment reuse keeps the L2 W stream (~1.3GB) from
// dominating over the HBM floor; each B fragment feeds two WMMAs.
// ---------------------------------------------------------------------------

typedef __attribute__((ext_vector_type(16))) _Float16 v16h;
typedef __attribute__((ext_vector_type(8)))  _Float16 v8h;
typedef __attribute__((ext_vector_type(8)))  float    v8f;

#define GROUPS    2
#define NUM_VARS  320
#define NCOLS     640            // GROUPS*NUM_VARS
#define CDIM      512
#define NROWS     65536          // 16*4096
#define CB_DIM    128
#define EPSV      1e-7f

#define M_TILE    32             // rows per block (2 WMMA M-subtiles)
#define NT_TOTAL  40             // 640/16 column tiles
#define KSTEPS    16             // 512/32 k-steps
#define A_STRIDE  520            // padded f16 row stride (bank-conflict free)
#define L_STRIDE  652            // padded f32 logits row stride

// --------------------------------------------------------------------------
// Kernel 0: zero the global accumulators (must be done every launch).
// --------------------------------------------------------------------------
__global__ void gvq_zero_kernel(int* __restrict__ g_counts,
                                float* __restrict__ g_probs) {
  for (int i = threadIdx.x; i < NCOLS; i += 256) {
    g_counts[i] = 0;
    g_probs[i]  = 0.0f;
  }
}

// --------------------------------------------------------------------------
// Kernel 1: convert W (512x640 f32, row-major) into f16 arranged in WMMA
// B-fragment order: Wh[((nt*16 + kstep)*32 + lane)*16 + i].
// Lane layout mirrors the 16-bit A/B fragment spec:
//   lanes 0-15 : N = lane,    i<8 -> K = kb+i,      i>=8 -> K = kb+16+(i-8)
//   lanes16-31 : N = lane-16, kb += 8 (same pattern)
// --------------------------------------------------------------------------
__global__ void gvq_convert_w_kernel(const float* __restrict__ W,
                                     _Float16* __restrict__ Wh) {
  int e = blockIdx.x * 256 + threadIdx.x;   // e < 512*640 = 327680
  int i    = e & 15;
  int lane = (e >> 4) & 31;
  int kst  = (e >> 9) & 15;
  int nt   = e >> 13;
  int n  = nt * 16 + (lane & 15);
  int kb = kst * 32 + ((lane & 16) ? 8 : 0);
  int k  = (i < 8) ? (kb + i) : (kb + 16 + (i - 8));
  Wh[e] = (_Float16)W[k * NCOLS + n];
}

// --------------------------------------------------------------------------
// Kernel 2: fused GEMM + argmax + softmax-accumulate + codebook gather.
// One block = 32 rows of x, 256 threads = 8 wave32s. Each wave owns 5 of the
// 40 column tiles and both 16-row M-subtiles; each B fragment feeds 2 WMMAs.
// --------------------------------------------------------------------------
__global__ void __launch_bounds__(256)
gvq_main_kernel(const float* __restrict__ x,
                const _Float16* __restrict__ Wh,
                const float* __restrict__ bias,
                const float* __restrict__ codebook,
                float* __restrict__ out,
                int* __restrict__ g_counts,
                float* __restrict__ g_probs) {
  __shared__ _Float16 sA[M_TILE * A_STRIDE];      // 32.5 KB f16 x-tile
  __shared__ float    sLogits[M_TILE * L_STRIDE]; // 81.5 KB logits tile
  __shared__ float    sProbs[NCOLS];
  __shared__ int      sCounts[NCOLS];
  __shared__ int      sK[M_TILE * GROUPS];

  const int tid  = threadIdx.x;
  const int wave = tid >> 5;
  const int lane = tid & 31;
  const int m0   = blockIdx.x * M_TILE;

  for (int i = tid; i < NCOLS; i += 256) {
    sProbs[i]  = 0.0f;
    sCounts[i] = 0;
  }

  // ---- stage x tile: 32 rows x 512 f32 -> f16 in LDS ---------------------
  {
    int row = tid >> 3;             // 0..31
    int c0  = (tid & 7) * 64;       // 8 threads cover one 512-wide row
    const float4* src = (const float4*)(x + (size_t)(m0 + row) * CDIM + c0);
    _Float16* dst = sA + row * A_STRIDE + c0;
#pragma unroll
    for (int j = 0; j < 16; ++j) {
      float4 v = src[j];
      dst[j * 4 + 0] = (_Float16)v.x;
      dst[j * 4 + 1] = (_Float16)v.y;
      dst[j * 4 + 2] = (_Float16)v.z;
      dst[j * 4 + 3] = (_Float16)v.w;
    }
  }
  __syncthreads();

  // ---- WMMA GEMM: 5 N-tiles x 2 M-subtiles per wave, 16 k-steps ----------
  v8f acc0[5] = {};                 // rows m0 .. m0+15
  v8f acc1[5] = {};                 // rows m0+16 .. m0+31
  const int rowA  = lane & 15;
  const int hiSel = (lane & 16) ? 8 : 0;   // lanes 16-31 hold K+8 / K+24

  // Per-lane B base: all (j, ks) offsets become compile-time immediates /
  // a single 1KB pointer increment (no 64-bit VALU chains in the loop).
  const _Float16* bbase =
      Wh + (size_t)wave * (5 * KSTEPS * 512) + lane * 16;

  for (int ks = 0; ks < KSTEPS; ++ks) {
    const _Float16* a0p = sA + rowA * A_STRIDE + ks * 32 + hiSel;
    const _Float16* a1p = a0p + 16 * A_STRIDE;
    v8h a0lo = *(const v8h*)(a0p);
    v8h a0hi = *(const v8h*)(a0p + 16);
    v8h a1lo = *(const v8h*)(a1p);
    v8h a1hi = *(const v8h*)(a1p + 16);
    v16h a0 = __builtin_shufflevector(a0lo, a0hi,
        0, 1, 2, 3, 4, 5, 6, 7, 8, 9, 10, 11, 12, 13, 14, 15);
    v16h a1 = __builtin_shufflevector(a1lo, a1hi,
        0, 1, 2, 3, 4, 5, 6, 7, 8, 9, 10, 11, 12, 13, 14, 15);

    const _Float16* bk = bbase + ks * 512;   // advances 1024B per k-step
#pragma unroll
    for (int j = 0; j < 5; ++j) {
      v16h bfrag = *(const v16h*)(bk + j * (KSTEPS * 512));  // imm offset
      acc0[j] = __builtin_amdgcn_wmma_f32_16x16x32_f16(
          false, a0, false, bfrag, (short)0, acc0[j], false, false);
      acc1[j] = __builtin_amdgcn_wmma_f32_16x16x32_f16(
          false, a1, false, bfrag, (short)0, acc1[j], false, false);
    }
  }

  // ---- bias add + scatter logits into LDS --------------------------------
  {
    const int rbase = (lane >> 4) * 8;      // C/D layout: lanes16-31 -> M+8
#pragma unroll
    for (int j = 0; j < 5; ++j) {
      int nt  = wave * 5 + j;
      int col = nt * 16 + (lane & 15);
      float bv = bias[col];
#pragma unroll
      for (int r = 0; r < 8; ++r) {
        sLogits[(rbase + r) * L_STRIDE + col]        = acc0[j][r] + bv;
        sLogits[(rbase + r + 16) * L_STRIDE + col]   = acc1[j][r] + bv;
      }
    }
  }
  __syncthreads();

  // ---- per-(row, group) argmax + softmax: one wave per 8 pairs ----------
  for (int q = 0; q < 8; ++q) {
    int pp  = wave * 8 + q;                 // 0..63 pairs
    int row = pp & 31;
    int g   = pp >> 5;
    const float* lrow = sLogits + row * L_STRIDE + g * NUM_VARS;

    float m = -1e30f;
    int   mi = 0;
#pragma unroll
    for (int i = 0; i < 10; ++i) {          // 320 cols / 32 lanes
      float v = lrow[lane + 32 * i];
      if (v > m) { m = v; mi = lane + 32 * i; }
    }
#pragma unroll
    for (int off = 16; off > 0; off >>= 1) {
      float om = __shfl_xor(m, off, 32);
      int   oi = __shfl_xor(mi, off, 32);
      if (om > m || (om == m && oi < mi)) { m = om; mi = oi; }
    }

    float ev[10];
    float s = 0.0f;
#pragma unroll
    for (int i = 0; i < 10; ++i) {
      ev[i] = __expf(lrow[lane + 32 * i] - m);
      s += ev[i];
    }
#pragma unroll
    for (int off = 16; off > 0; off >>= 1) s += __shfl_xor(s, off, 32);
    float inv = 1.0f / s;
#pragma unroll
    for (int i = 0; i < 10; ++i)
      atomicAdd(&sProbs[g * NUM_VARS + lane + 32 * i], ev[i] * inv);

    if (lane == 0) {
      sK[row * GROUPS + g] = mi;
      atomicAdd(&sCounts[g * NUM_VARS + mi], 1);
    }
  }
  __syncthreads();

  // ---- flush block-local accumulators to global --------------------------
  for (int i = tid; i < NCOLS; i += 256) {
    int c = sCounts[i];
    if (c) atomicAdd(&g_counts[i], c);
    atomicAdd(&g_probs[i], sProbs[i]);
  }

  // ---- codebook gather: out[row, g*128+j] = codebook[g*320+k, j] ---------
  for (int idx = tid; idx < M_TILE * GROUPS * CB_DIM; idx += 256) {
    int row = idx >> 8;                     // /256
    int rem = idx & 255;
    int g   = rem >> 7;
    int j   = rem & 127;
    int k   = sK[row * GROUPS + g];
    out[(size_t)(m0 + row) * (GROUPS * CB_DIM) + rem] =
        codebook[(size_t)(g * NUM_VARS + k) * CB_DIM + j];
  }
}

// --------------------------------------------------------------------------
// Kernel 3: perplexities from the global accumulators (one tiny block).
// --------------------------------------------------------------------------
__global__ void gvq_finalize_kernel(const int* __restrict__ g_counts,
                                    const float* __restrict__ g_probs,
                                    float* __restrict__ out_scalars) {
  __shared__ float sh[GROUPS], ss[GROUPS];
  int t = threadIdx.x;                      // 64 threads = 2 wave32s
  int g = t >> 5, lane = t & 31;
  const float invN = 1.0f / (float)NROWS;

  float hsum = 0.0f, psum = 0.0f;
#pragma unroll
  for (int i = 0; i < 10; ++i) {
    int v = lane + 32 * i;
    float ph = (float)g_counts[g * NUM_VARS + v] * invN;
    hsum += ph * __logf(ph + EPSV);
    float ps = g_probs[g * NUM_VARS + v] * invN;
    psum += ps * __logf(ps + EPSV);
  }
#pragma unroll
  for (int off = 16; off > 0; off >>= 1) {
    hsum += __shfl_xor(hsum, off, 32);
    psum += __shfl_xor(psum, off, 32);
  }
  if (lane == 0) { sh[g] = __expf(-hsum); ss[g] = __expf(-psum); }
  __syncthreads();
  if (t == 0) {
    out_scalars[0] = sh[0] + sh[1];   // code_perplexity
    out_scalars[1] = ss[0] + ss[1];   // prob_perplexity
  }
}

// --------------------------------------------------------------------------
extern "C" void kernel_launch(void* const* d_in, const int* in_sizes, int n_in,
                              void* d_out, int out_size, void* d_ws,
                              size_t ws_size, hipStream_t stream) {
  const float* x  = (const float*)d_in[0];   // (16,4096,512)
  const float* W  = (const float*)d_in[1];   // (512,640)
  const float* b  = (const float*)d_in[2];   // (640,)
  const float* cb = (const float*)d_in[3];   // (640,128)
  float* out = (float*)d_out;                // 65536*256 + 2 floats

  // workspace layout
  char* ws = (char*)d_ws;
  _Float16* Wh      = (_Float16*)ws;                      // 655360 B
  int*      gcounts = (int*)(ws + 655360);                // 2560 B
  float*    gprobs  = (float*)(ws + 655360 + NCOLS * 4);  // 2560 B

  gvq_zero_kernel<<<1, 256, 0, stream>>>(gcounts, gprobs);
  gvq_convert_w_kernel<<<(CDIM * NCOLS) / 256, 256, 0, stream>>>(W, Wh);
  gvq_main_kernel<<<NROWS / M_TILE, 256, 0, stream>>>(
      x, Wh, b, cb, out, gcounts, gprobs);
  gvq_finalize_kernel<<<1, 64, 0, stream>>>(
      gcounts, gprobs, out + (size_t)NROWS * GROUPS * CB_DIM);
}
